// microcircuit_59588376264833
// MI455X (gfx1250) — compile-verified, matrix operand order checked by reading, not run on hardware
//
#include <hip/hip_runtime.h>
#include <hip/hip_bf16.h>

// ---------------- problem constants ----------------
#define B_  256
#define T_  128
#define NF_ 512

#define AS1 __attribute__((address_space(1)))

typedef __bf16 bf16_t;
typedef __attribute__((ext_vector_type(16))) __bf16 v16bf;
typedef __attribute__((ext_vector_type(8)))  __bf16 v8bf;
typedef __attribute__((ext_vector_type(8)))  float  v8f;
typedef __attribute__((ext_vector_type(4)))  float  f32x4;

// ---------------- stage kernel arg structs (shared host/device) ----------------
struct Term { const void* A; const void* W; int lda; int K; int a_f32; };
struct Job  { Term t[3]; const float* add; void* out_bf; float* out_f;
              int M, N, nt, ldo, ldo_f, tanh_en; };
struct StageArgs { Job j[2]; int njobs; };

// generic -> global(AS1) pointer (bit-identical for global memory on amdgcn)
template <typename T>
__device__ inline const AS1 T* gcp(const void* p) {
  return (const AS1 T*)(unsigned long long)p;
}
template <typename T>
__device__ inline AS1 T* gmp(void* p) {
  return (AS1 T*)(unsigned long long)p;
}

// ---------------- fragment loaders (global address space) ----------------
// CDNA5 16-bit A-matrix 16x32 layout: lane L holds row m=L%16; K-base kb=(L/16)*8;
// elements 0..7 -> K = kb..kb+7, elements 8..15 -> K = kb+16..kb+23.
// B (32x16) with W stored [N,K] row-major uses the identical pattern with row = n.
// `p` is pre-offset to row*ld + kb.
__device__ inline v16bf ldfrag_bfg(const AS1 bf16_t* p) {
  union { v16bf v; v8bf h[2]; } u;
  u.h[0] = *(const AS1 v8bf*)(p);
  u.h[1] = *(const AS1 v8bf*)(p + 16);
  return u.v;
}

__device__ inline v16bf ldfrag_f32g(const AS1 float* p) {
  f32x4 a0 = *(const AS1 f32x4*)(p);
  f32x4 a1 = *(const AS1 f32x4*)(p + 4);
  f32x4 b0 = *(const AS1 f32x4*)(p + 16);
  f32x4 b1 = *(const AS1 f32x4*)(p + 20);
  v16bf v;
  v[0]=(bf16_t)a0.x;  v[1]=(bf16_t)a0.y;  v[2]=(bf16_t)a0.z;  v[3]=(bf16_t)a0.w;
  v[4]=(bf16_t)a1.x;  v[5]=(bf16_t)a1.y;  v[6]=(bf16_t)a1.z;  v[7]=(bf16_t)a1.w;
  v[8]=(bf16_t)b0.x;  v[9]=(bf16_t)b0.y;  v[10]=(bf16_t)b0.z; v[11]=(bf16_t)b0.w;
  v[12]=(bf16_t)b1.x; v[13]=(bf16_t)b1.y; v[14]=(bf16_t)b1.z; v[15]=(bf16_t)b1.w;
  return v;
}

__device__ inline v8f wmma_bf16(v16bf a, v16bf b, v8f c) {
  return __builtin_amdgcn_wmma_f32_16x16x32_bf16(false, a, false, b, (short)0, c,
                                                 false, false);
}

// hardware V_TANH_F32 (gfx1250 trans op) when the builtin exists
#if __has_builtin(__builtin_amdgcn_tanhf)
__device__ inline float fast_tanh(float x) { return __builtin_amdgcn_tanhf(x); }
#else
__device__ inline float fast_tanh(float x) { return tanhf(x); }
#endif

// ---------------- main fused GEMM(+GEMM+GEMM)+tanh stage kernel ----------------
// out[M,N] = act( sum_terms A_i[M,K_i] @ W_i[N,K_i]^T  (+ add[N]) )
// block = 256 threads = 8 wave32 waves; block tile 64x64; wave tile 16x32.
__global__ __launch_bounds__(256) void stage_kernel(StageArgs sa) {
  int jid = blockIdx.z;
  if (jid >= sa.njobs) return;
  const Job J = sa.j[jid];
  int ntN = J.N >> 6;                    // N / 64
  int nTiles = (J.M >> 6) * ntN;
  if ((int)blockIdx.x >= nTiles) return;
  int bm = blockIdx.x / ntN, bn = blockIdx.x % ntN;

  int tid  = threadIdx.x;
  int wave = tid >> 5, lane = tid & 31;
  int m0 = bm * 64 + (wave & 3) * 16;    // 4 wave-rows
  int n0 = bn * 64 + (wave >> 2) * 32;   // 2 wave-cols of 32
  int kb = (lane >> 4) << 3;             // per-lane K base within 32-chunk
  int rA  = m0 + (lane & 15);
  int rB0 = n0 + (lane & 15);
  int rB1 = rB0 + 16;

  v8f acc0 = {0.f,0.f,0.f,0.f,0.f,0.f,0.f,0.f};
  v8f acc1 = acc0;

  for (int it = 0; it < J.nt; ++it) {
    const Term T = J.t[it];
    const int K = T.K;
    const AS1 bf16_t* w0 = gcp<bf16_t>(T.W) + (size_t)rB0 * K + kb;
    const AS1 bf16_t* w1 = gcp<bf16_t>(T.W) + (size_t)rB1 * K + kb;
    if (T.a_f32) {
      // f32 activation source (the raw x input), converted in-register
      const AS1 float* ap = gcp<float>(T.A) + (size_t)rA * T.lda + kb;
      for (int k0 = 0; k0 < K; k0 += 32) {
        v16bf a = ldfrag_f32g(ap + k0);
        acc0 = wmma_bf16(a, ldfrag_bfg(w0 + k0), acc0);
        acc1 = wmma_bf16(a, ldfrag_bfg(w1 + k0), acc1);
      }
    } else {
      // bf16 path, K unrolled by 64 (all K are multiples of 64): 4 WMMAs / iter
      const AS1 bf16_t* ap = gcp<bf16_t>(T.A) + (size_t)rA * T.lda + kb;
      for (int k0 = 0; k0 < K; k0 += 64) {
        v16bf a0  = ldfrag_bfg(ap + k0);
        v16bf a1  = ldfrag_bfg(ap + k0 + 32);
        v16bf b00 = ldfrag_bfg(w0 + k0);
        v16bf b10 = ldfrag_bfg(w1 + k0);
        v16bf b01 = ldfrag_bfg(w0 + k0 + 32);
        v16bf b11 = ldfrag_bfg(w1 + k0 + 32);
        acc0 = wmma_bf16(a0, b00, acc0);
        acc1 = wmma_bf16(a0, b10, acc1);
        acc0 = wmma_bf16(a1, b01, acc0);
        acc1 = wmma_bf16(a1, b11, acc1);
      }
    }
  }

  int col0 = n0 + (lane & 15), col1 = col0 + 16;
  if (J.add) {
    const AS1 float* av = gcp<float>(J.add);
    float a0 = av[col0], a1 = av[col1];
    for (int g = 0; g < 8; ++g) { acc0[g] += a0; acc1[g] += a1; }
  }
  int rbase = m0 + ((lane >> 4) << 3);   // C/D layout: reg g -> M = g (+8 upper half)
  AS1 bf16_t* ob = gmp<bf16_t>(J.out_bf);
  AS1 float*  of = gmp<float>(J.out_f);
  for (int g = 0; g < 8; ++g) {
    float v0 = acc0[g], v1 = acc1[g];
    if (J.tanh_en) { v0 = fast_tanh(v0); v1 = fast_tanh(v1); }
    int m = rbase + g;
    if (J.out_bf) {
      ob[(size_t)m * J.ldo + col0] = (bf16_t)v0;
      ob[(size_t)m * J.ldo + col1] = (bf16_t)v1;
    }
    if (J.out_f) {
      of[(size_t)m * J.ldo_f + col0] = v0;
      of[(size_t)m * J.ldo_f + col1] = v1;
    }
  }
}

// ---------------- prep kernels (run every launch; deterministic) ----------------
// f32 slice -> bf16 [N,K] row-major
__global__ void k_convert(const float* src_, int srcld, int coff,
                          __hip_bfloat16* dst_, int total, int K) {
  int i = blockIdx.x * blockDim.x + threadIdx.x;
  if (i >= total) return;
  const AS1 float* src = gcp<float>(src_);
  AS1 bf16_t* dst = gmp<bf16_t>(dst_);
  int n = i / K, k = i - n * K;
  dst[i] = (bf16_t)src[(size_t)n * srcld + coff + k];
}

// fused[n,k] = sum_j Wih[n, coff+j] * (Wlin[j,k] * mask[j,k])   -> bf16 [N,K]
__global__ void k_fuse(const float* wih_, int wihld, int coff,
                       const float* wl_, const float* msk_,
                       __hip_bfloat16* dst_, int N, int Jd, int K) {
  int i = blockIdx.x * blockDim.x + threadIdx.x;
  if (i >= N * K) return;
  const AS1 float* wih = gcp<float>(wih_);
  const AS1 float* wl  = gcp<float>(wl_);
  const AS1 float* msk = gcp<float>(msk_);
  AS1 bf16_t* dst = gmp<bf16_t>(dst_);
  int n = i / K, k = i - n * K;
  const AS1 float* wrow = wih + (size_t)n * wihld + coff;
  float s = 0.f;
  for (int j = 0; j < Jd; ++j)
    s += wrow[j] * wl[(size_t)j * K + k] * msk[(size_t)j * K + k];
  dst[i] = (bf16_t)s;
}

// FB[t,n] = sum_j Wih_H56[n, 384+j] * proj[t,j]   (f32, 128x384)
__global__ void k_fb(const float* wih_, const float* proj_, float* fb_) {
  int i = blockIdx.x * blockDim.x + threadIdx.x;
  if (i >= T_ * 384) return;
  const AS1 float* wih  = gcp<float>(wih_);
  const AS1 float* proj = gcp<float>(proj_);
  AS1 float* fb = gmp<float>(fb_);
  int t = i / 384, n = i - t * 384;
  float s = 0.f;
  for (int j = 0; j < 384; ++j)
    s += wih[(size_t)n * 768 + 384 + j] * proj[(size_t)t * 384 + j];
  fb[i] = s;
}

// elementwise f32 -> bf16
__global__ void k_cvt1d(const float* src_, __hip_bfloat16* dst_, int n) {
  int i = blockIdx.x * blockDim.x + threadIdx.x;
  if (i >= n) return;
  gmp<bf16_t>(dst_)[i] = (bf16_t)gcp<float>(src_)[i];
}

// ---------------- host ----------------
static inline int cdiv(int a, int b) { return (a + b - 1) / b; }

extern "C" void kernel_launch(void* const* d_in, const int* in_sizes, int n_in,
                              void* d_out, int out_size, void* d_ws, size_t ws_size,
                              hipStream_t stream) {
  (void)in_sizes; (void)n_in; (void)out_size; (void)ws_size;
  // ---- input index map (depth-first dict insertion order of setup_inputs) ----
  // 0: x [B,T,NF]
  // params: 9 RNN cells (L4,L23,L56,I4,I23,I56,H4,H23,H56) x {Wih,Whh,bih,bhh} -> 1..36
  // 15 linears (WFF1,WFF2,WFBa1,WFBb1,WFBa2,WFBb2,WL423,WL2356,WL564,
  //             WI423,WI2356,WI564,WH423,WH2356,WH564) x {W,b}            -> 37..66
  // opLayer {W,b} -> 67,68 ; proj -> 69 ; h0 (9, RNN order) -> 70..78 ; masks (15) -> 79..93
  const float* x = (const float*)d_in[0];
  auto F32 = [&](int i) { return (const float*)d_in[i]; };

  static const int dh[9]    = {256, 512, 384, 256, 512, 384, 256, 512, 384};
  static const int wihld[9] = {768, 1024, 768, 512, 1024, 768, 512, 512, 768};
  auto WIH = [&](int c) { return 1 + 4 * c; };
  auto WHH = [&](int c) { return 2 + 4 * c; };
  auto H0  = [&](int c) { return 70 + c; };

  // ---- workspace allocator (256B aligned) ----
  char* base = (char*)d_ws;
  size_t off = 0;
  auto alloc = [&](size_t bytes) -> void* {
    void* p = base + off;
    off += (bytes + 255) & ~(size_t)255;
    return p;
  };

  __hip_bfloat16* Whh_bf[9];
  __hip_bfloat16* hbuf[9][2];
  for (int c = 0; c < 9; ++c) {
    Whh_bf[c]  = (__hip_bfloat16*)alloc((size_t)dh[c] * dh[c] * 2);
    hbuf[c][0] = (__hip_bfloat16*)alloc((size_t)B_ * dh[c] * 2);
    hbuf[c][1] = (__hip_bfloat16*)alloc((size_t)B_ * dh[c] * 2);
  }
  __hip_bfloat16* A_L4x = (__hip_bfloat16*)alloc((size_t)256 * 512 * 2);
  __hip_bfloat16* Wop   = (__hip_bfloat16*)alloc((size_t)512 * 512 * 2);
  float* FBv            = (float*)alloc((size_t)T_ * 384 * 4);

  // fused matrices: {cell, coff, linW idx, mask idx, N, J, K}
  struct FD { int cell, coff, wi, mi, N, J, K; };
  static const FD fd[15] = {
    {0, 512, 53, 87, 256, 256, 384},  // L4u  <- WL564
    {1,   0, 49, 85, 512, 512, 256},  // L23a <- WL423
    {1, 512, 41, 81, 512, 512, 384},  // L23b <- WFBa1
    {2,   0, 51, 86, 384, 384, 512},  // L56a <- WL2356
    {2, 384, 43, 82, 384, 384, 384},  // L56b <- WFBb1
    {3,   0, 37, 79, 256, 256, 512},  // I4a  <- WFF1
    {3, 256, 59, 90, 256, 256, 384},  // I4b  <- WI564
    {4,   0, 55, 88, 512, 512, 256},  // I23a <- WI423
    {4, 512, 45, 83, 512, 512, 384},  // I23b <- WFBa2
    {5,   0, 57, 89, 384, 384, 512},  // I56a <- WI2356
    {5, 384, 47, 84, 384, 384, 384},  // I56b <- WFBb2
    {6,   0, 39, 80, 256, 256, 512},  // H4a  <- WFF2
    {6, 256, 65, 93, 256, 256, 384},  // H4b  <- WH564
    {7,   0, 61, 91, 512, 512, 256},  // H23a <- WH423
    {8,   0, 63, 92, 384, 384, 512},  // H56a <- WH2356
  };
  __hip_bfloat16* Fm[15];
  for (int i = 0; i < 15; ++i)
    Fm[i] = (__hip_bfloat16*)alloc((size_t)fd[i].N * fd[i].K * 2);

  // ---- prep: weight conversion / fusion ----
  for (int c = 0; c < 9; ++c) {
    int tot = dh[c] * dh[c];
    k_convert<<<cdiv(tot, 256), 256, 0, stream>>>(F32(WHH(c)), dh[c], 0, Whh_bf[c], tot, dh[c]);
    int ne = B_ * dh[c];
    k_cvt1d<<<cdiv(ne, 256), 256, 0, stream>>>(F32(H0(c)), hbuf[c][0], ne);
  }
  k_convert<<<cdiv(256 * 512, 256), 256, 0, stream>>>(F32(WIH(0)), 768, 0, A_L4x, 256 * 512, 512);
  k_convert<<<cdiv(512 * 512, 256), 256, 0, stream>>>(F32(67), 512, 0, Wop, 512 * 512, 512);
  for (int i = 0; i < 15; ++i) {
    const FD& f = fd[i];
    k_fuse<<<cdiv(f.N * f.K, 256), 256, 0, stream>>>(
        F32(WIH(f.cell)), wihld[f.cell], f.coff, F32(f.wi), F32(f.mi), Fm[i], f.N, f.J, f.K);
  }
  k_fb<<<cdiv(T_ * 384, 256), 256, 0, stream>>>(F32(WIH(8)), F32(69), FBv);

  // ---- stage launch helpers ----
  auto mkT = [](const void* A, int lda, int af32, const void* W, int K) {
    Term t; t.A = A; t.W = W; t.lda = lda; t.K = K; t.a_f32 = af32; return t;
  };
  auto mkJ = [](int N, Term t0, Term t1, Term t2, int nt, const float* add,
                void* ob, int ldo, float* of, int ldo_f, int tanh_en) {
    Job j; j.t[0] = t0; j.t[1] = t1; j.t[2] = t2; j.add = add;
    j.out_bf = ob; j.out_f = of; j.M = B_; j.N = N; j.nt = nt;
    j.ldo = ldo; j.ldo_f = ldo_f; j.tanh_en = tanh_en; return j;
  };
  Term tz{}; // unused slot
  auto launch1 = [&](const Job& j0) {
    StageArgs sa{}; sa.j[0] = j0; sa.njobs = 1;
    int tiles = (j0.M / 64) * (j0.N / 64);
    stage_kernel<<<dim3(tiles, 1, 1), dim3(256), 0, stream>>>(sa);
  };
  auto launch2 = [&](const Job& j0, const Job& j1) {
    StageArgs sa{}; sa.j[0] = j0; sa.j[1] = j1; sa.njobs = 2;
    int t0 = (j0.M / 64) * (j0.N / 64), t1 = (j1.M / 64) * (j1.N / 64);
    stage_kernel<<<dim3(t0 > t1 ? t0 : t1, 1, 2), dim3(256), 0, stream>>>(sa);
  };

  float* out = (float*)d_out;

  // ---- 128 time steps, 7 dependent stage launches each ----
  for (int t = 0; t < T_; ++t) {
    int cur = t & 1, nxt = cur ^ 1;

    // L4 = tanh(x_t @ A_L4x^T + hL56 @ L4u^T + hL4 @ Whh^T)
    launch1(mkJ(256,
        mkT(x + (size_t)t * NF_, T_ * NF_, 1, A_L4x, 512),
        mkT(hbuf[2][cur], 384, 0, Fm[0], 384),
        mkT(hbuf[0][cur], 256, 0, Whh_bf[0], 256),
        3, nullptr, hbuf[0][nxt], 256, nullptr, 0, 1));

    // L23
    launch1(mkJ(512,
        mkT(hbuf[0][nxt], 256, 0, Fm[1], 256),
        mkT(hbuf[5][cur], 384, 0, Fm[2], 384),
        mkT(hbuf[1][cur], 512, 0, Whh_bf[1], 512),
        3, nullptr, hbuf[1][nxt], 512, nullptr, 0, 1));

    // L56 || I4
    launch2(
      mkJ(384,
        mkT(hbuf[1][nxt], 512, 0, Fm[3], 512),
        mkT(hbuf[5][cur], 384, 0, Fm[4], 384),
        mkT(hbuf[2][cur], 384, 0, Whh_bf[2], 384),
        3, nullptr, hbuf[2][nxt], 384, nullptr, 0, 1),
      mkJ(256,
        mkT(hbuf[1][nxt], 512, 0, Fm[5], 512),
        mkT(hbuf[5][cur], 384, 0, Fm[6], 384),
        mkT(hbuf[3][cur], 256, 0, Whh_bf[3], 256),
        3, nullptr, hbuf[3][nxt], 256, nullptr, 0, 1));

    // I23
    launch1(mkJ(512,
        mkT(hbuf[3][nxt], 256, 0, Fm[7], 256),
        mkT(hbuf[8][cur], 384, 0, Fm[8], 384),
        mkT(hbuf[4][cur], 512, 0, Whh_bf[4], 512),
        3, nullptr, hbuf[4][nxt], 512, nullptr, 0, 1));

    // I56 || H4
    launch2(
      mkJ(384,
        mkT(hbuf[4][nxt], 512, 0, Fm[9], 512),
        mkT(hbuf[8][cur], 384, 0, Fm[10], 384),
        mkT(hbuf[5][cur], 384, 0, Whh_bf[5], 384),
        3, nullptr, hbuf[5][nxt], 384, nullptr, 0, 1),
      mkJ(256,
        mkT(hbuf[4][nxt], 512, 0, Fm[11], 512),
        mkT(hbuf[8][cur], 384, 0, Fm[12], 384),
        mkT(hbuf[6][cur], 256, 0, Whh_bf[6], 256),
        3, nullptr, hbuf[6][nxt], 256, nullptr, 0, 1));

    // H23
    launch1(mkJ(512,
        mkT(hbuf[6][nxt], 256, 0, Fm[13], 256),
        mkT(hbuf[7][cur], 512, 0, Whh_bf[7], 512),
        tz, 2, nullptr, hbuf[7][nxt], 512, nullptr, 0, 1));

    // H56 (+FB[t]) || pred (f32 to d_out, no tanh)
    launch2(
      mkJ(384,
        mkT(hbuf[7][nxt], 512, 0, Fm[14], 512),
        mkT(hbuf[8][cur], 384, 0, Whh_bf[8], 384),
        tz, 2, FBv + (size_t)t * 384, hbuf[8][nxt], 384, nullptr, 0, 1),
      mkJ(512,
        mkT(hbuf[7][nxt], 512, 0, Wop, 512),
        tz, tz, 1, nullptr, nullptr, 0, out + (size_t)t * NF_, T_ * NF_, 0));
  }
}
